// Net_25469156065595
// MI455X (gfx1250) — matrix operand order repaired and checked by reading, not compiled, not run on hardware
//
#include <hip/hip_runtime.h>
#include <hip/hip_bf16.h>

// ---- problem constants ----
#define BB 32
#define TT 2048
#define HH 512
#define CC 64
#define RR 200
#define RPAD 208           // 13 * 16

typedef __attribute__((ext_vector_type(16))) __bf16 v16bf;
typedef __attribute__((ext_vector_type(8)))  __bf16 v8bf;
typedef __attribute__((ext_vector_type(4)))  __bf16 v4bf;
typedef __attribute__((ext_vector_type(8)))  float  v8f;

#define WMMA_BF16(a, b, c) \
    __builtin_amdgcn_wmma_f32_16x16x32_bf16(false, (a), false, (b), (short)0, (c), false, false)

// =====================================================================
// Kernel 1: atn_logit = seq @ W_attn + b ; exp+mask ; colsum atomics
//   grid (T/128, B), block 256 (8 waves). M=128, N=64, K=512 (chunks of 32)
// =====================================================================
__global__ __launch_bounds__(256)
void k1_gemm_exp(const float* __restrict__ seq, const int* __restrict__ seq_len,
                 const float* __restrict__ W, const float* __restrict__ bias,
                 float* __restrict__ expl, float* __restrict__ colsum)
{
    __shared__ __bf16 As[128 * 48];   // seq tile  [m=128][k=32], stride 48 halves
    __shared__ __bf16 Ws[64 * 48];    // W^T tile  [c=64][k=32]

    const int b    = blockIdx.y;
    const int t0   = blockIdx.x * 128;
    const int tid  = threadIdx.x;
    const int wave = tid >> 5;
    const int lane = tid & 31;
    const int len  = seq_len[b];

    v8f acc[4] = {};

    for (int kc = 0; kc < HH; kc += 32) {
        __syncthreads();
        // stage seq tile: 128 x 32 fp32 -> bf16 (row major)
        for (int q = tid; q < 1024; q += 256) {
            int row = q >> 3;
            int c4  = (q & 7) * 4;
            float4 v = *(const float4*)(seq + ((size_t)b * TT + t0 + row) * HH + kc + c4);
            v4bf o = { (__bf16)v.x, (__bf16)v.y, (__bf16)v.z, (__bf16)v.w };
            *(v4bf*)(&As[row * 48 + c4]) = o;
        }
        // stage W chunk transposed: W[kc+k][c] -> Ws[c][k]
        for (int q = tid; q < 512; q += 256) {
            int k  = q >> 4;
            int c4 = (q & 15) * 4;
            float4 v = *(const float4*)(W + (size_t)(kc + k) * CC + c4);
            Ws[(c4 + 0) * 48 + k] = (__bf16)v.x;
            Ws[(c4 + 1) * 48 + k] = (__bf16)v.y;
            Ws[(c4 + 2) * 48 + k] = (__bf16)v.z;
            Ws[(c4 + 3) * 48 + k] = (__bf16)v.w;
        }
        __syncthreads();

        // A fragment (16x32 bf16): lanes 0-15 K {0..7,16..23}, lanes 16-31 +8
        int m    = wave * 16 + (lane & 15);
        int koff = (lane & 16) ? 8 : 0;
        v8bf alo = *(const v8bf*)(&As[m * 48 + koff]);
        v8bf ahi = *(const v8bf*)(&As[m * 48 + 16 + koff]);
        v16bf a  = __builtin_shufflevector(alo, ahi,
                       0,1,2,3,4,5,6,7,8,9,10,11,12,13,14,15);
        int boff = (lane & 16) ? 16 : 0;
        #pragma unroll
        for (int n = 0; n < 4; ++n) {
            int cn = n * 16 + (lane & 15);
            v8bf blo = *(const v8bf*)(&Ws[cn * 48 + koff]);
            v8bf bhi = *(const v8bf*)(&Ws[cn * 48 + 16 + koff]);
            v16bf bb = __builtin_shufflevector(blo, bhi, 0,1,2,3,4,5,6,7,8,9,10,11,12,13,14,15);
            acc[n] = WMMA_BF16(a, bb, acc[n]);
        }
    }

    // epilogue: bias, exp, mask, store exp_logit, column-sum atomics
    #pragma unroll
    for (int n = 0; n < 4; ++n) {
        int   c  = n * 16 + (lane & 15);
        float bi = bias[c];
        float s  = 0.f;
        #pragma unroll
        for (int r = 0; r < 8; ++r) {
            int t = t0 + wave * 16 + r + ((lane & 16) ? 8 : 0);
            float e = (t < len) ? __expf(acc[n][r] + bi) : 0.f;
            expl[((size_t)b * TT + t) * CC + c] = e;
            s += e;
        }
        atomicAdd(&colsum[b * CC + c], s);
    }
}

// =====================================================================
// Kernel 2: temporal_attention = exp_logit / colsum   (vectorized x4)
// =====================================================================
__global__ __launch_bounds__(256)
void k2_normalize(const float* __restrict__ expl, const float* __restrict__ colsum,
                  float* __restrict__ ta)
{
    size_t i4 = ((size_t)blockIdx.x * 256 + threadIdx.x) * 4;
    int b  = (int)(i4 >> 17);          // / (T*C)
    int c0 = (int)(i4 & 63);
    float4 e = *(const float4*)(expl + i4);
    float4 s = *(const float4*)(colsum + b * CC + c0);
    float4 o = { e.x / s.x, e.y / s.y, e.z / s.z, e.w / s.w };
    *(float4*)(ta + i4) = o;
}

// =====================================================================
// Kernel 3: action_feature[b] = (exp_logit^T @ seq) / colsum
//   grid (B), block 256. M=64 (C), N=512 (H), K=2048 (T, chunks of 32)
//   wave w: mtile = w&3, N-half = (w>>2)*256 (16 n-tiles)
// =====================================================================
__global__ __launch_bounds__(256)
void k3_action_feature(const float* __restrict__ seq, const float* __restrict__ expl,
                       const float* __restrict__ colsum, float* __restrict__ af)
{
    __shared__ __bf16 At[64 * 48];    // exp_logit^T tile [c=64][k=32]
    __shared__ __bf16 Bs[512 * 48];   // seq^T tile       [h=512][k=32]

    const int b     = blockIdx.x;
    const int tid   = threadIdx.x;
    const int wave  = tid >> 5;
    const int lane  = tid & 31;
    const int mtile = wave & 3;
    const int nbase = (wave >> 2) * 256;

    v8f acc[16] = {};

    for (int kc = 0; kc < TT; kc += 32) {
        __syncthreads();
        // stage exp_logit tile transposed: expl[b][kc+k][c] -> At[c][k]
        for (int q = tid; q < 512; q += 256) {
            int k  = q >> 4;
            int c4 = (q & 15) * 4;
            float4 v = *(const float4*)(expl + ((size_t)b * TT + kc + k) * CC + c4);
            At[(c4 + 0) * 48 + k] = (__bf16)v.x;
            At[(c4 + 1) * 48 + k] = (__bf16)v.y;
            At[(c4 + 2) * 48 + k] = (__bf16)v.z;
            At[(c4 + 3) * 48 + k] = (__bf16)v.w;
        }
        // stage seq tile transposed: seq[b][kc+k][h] -> Bs[h][k]
        for (int q = tid; q < 4096; q += 256) {
            int k  = q >> 7;
            int h4 = (q & 127) * 4;
            float4 v = *(const float4*)(seq + ((size_t)b * TT + kc + k) * HH + h4);
            Bs[(h4 + 0) * 48 + k] = (__bf16)v.x;
            Bs[(h4 + 1) * 48 + k] = (__bf16)v.y;
            Bs[(h4 + 2) * 48 + k] = (__bf16)v.z;
            Bs[(h4 + 3) * 48 + k] = (__bf16)v.w;
        }
        __syncthreads();

        int m    = mtile * 16 + (lane & 15);
        int koff = (lane & 16) ? 8 : 0;
        v8bf alo = *(const v8bf*)(&At[m * 48 + koff]);
        v8bf ahi = *(const v8bf*)(&At[m * 48 + 16 + koff]);
        v16bf a  = __builtin_shufflevector(alo, ahi,
                       0,1,2,3,4,5,6,7,8,9,10,11,12,13,14,15);
        int boff = (lane & 16) ? 16 : 0;
        #pragma unroll
        for (int nt = 0; nt < 16; ++nt) {
            int h = nbase + nt * 16 + (lane & 15);
            v8bf blo = *(const v8bf*)(&Bs[h * 48 + koff]);
            v8bf bhi = *(const v8bf*)(&Bs[h * 48 + 16 + koff]);
            v16bf bb = __builtin_shufflevector(blo, bhi, 0,1,2,3,4,5,6,7,8,9,10,11,12,13,14,15);
            acc[nt] = WMMA_BF16(a, bb, acc[nt]);
        }
    }

    // epilogue: divide rows by colsum, store action_feature [C][H]
    float inv[8];
    #pragma unroll
    for (int r = 0; r < 8; ++r) {
        int c = mtile * 16 + r + ((lane & 16) ? 8 : 0);
        inv[r] = 1.f / colsum[b * CC + c];
    }
    #pragma unroll
    for (int nt = 0; nt < 16; ++nt) {
        int h = nbase + nt * 16 + (lane & 15);
        #pragma unroll
        for (int r = 0; r < 8; ++r) {
            int c = mtile * 16 + r + ((lane & 16) ? 8 : 0);
            af[((size_t)b * CC + c) * HH + h] = acc[nt][r] * inv[r];
        }
    }
}

// =====================================================================
// Kernel 4: recipe path + action_logit. grid (B), block 256, dynamic LDS.
//   P[0] = af @ recipe_attention_matrix^T (pre-softmax), P[1] = af @ recipe_matrix^T
//   recipe_logit = sum_c softmax_c(P0) * P1 + bias ; action_logit = rowdot(af, am)
// =====================================================================
#define AFS_STRIDE 528                         // halves, keeps 32B alignment
#define AFS_BYTES  (CC * AFS_STRIDE * 2)       // 67584
#define P_FLOATS   (2 * RPAD * CC)             // 26624

__global__ __launch_bounds__(256)
void k4_recipe(const float* __restrict__ af,
               const float* __restrict__ am,  const float* __restrict__ ab,
               const float* __restrict__ ram, const float* __restrict__ rm,
               const float* __restrict__ rbias,
               float* __restrict__ action_out, float* __restrict__ recipe_out)
{
    extern __shared__ char smem[];
    __bf16* afs = (__bf16*)smem;                       // [c=64][k=512] bf16
    float*  P   = (float*)(smem + AFS_BYTES);          // [2][208][64]

    const int b    = blockIdx.x;
    const int tid  = threadIdx.x;
    const int wave = tid >> 5;
    const int lane = tid & 31;

    // stage af[b] as bf16, layout already [N=c][K=h] row-major
    for (int q = tid; q < 8192; q += 256) {
        int row = q >> 7;
        int h4  = (q & 127) * 4;
        float4 v = *(const float4*)(af + ((size_t)b * CC + row) * HH + h4);
        v4bf o = { (__bf16)v.x, (__bf16)v.y, (__bf16)v.z, (__bf16)v.w };
        *(v4bf*)(&afs[row * AFS_STRIDE + h4]) = o;
    }
    __syncthreads();

    // two M=208,N=64,K=512 GEMMs: wave w -> gemm g = w>>2, ntile = w&3
    {
        const int g  = wave >> 2;
        const int nt = wave & 3;
        const float* Amat = g ? rm : ram;
        float* Pg = P + g * RPAD * CC;
        const int cn   = nt * 16 + (lane & 15);
        const int koffA = (lane & 16) ? 8 : 0;
        const int bsel  = (lane & 16) ? 16 : 0;

        for (int mt = 0; mt < 13; ++mt) {
            int rA = mt * 16 + (lane & 15);
            if (rA > RR - 1) rA = RR - 1;              // clamp OOB rows (ignored later)
            v8f acc = {};
            for (int kc = 0; kc < HH; kc += 32) {
                const float* ap = Amat + (size_t)rA * HH + kc + koffA;
                float4 f0 = *(const float4*)(ap);
                float4 f1 = *(const float4*)(ap + 4);
                float4 f2 = *(const float4*)(ap + 16);
                float4 f3 = *(const float4*)(ap + 20);
                v16bf a = { (__bf16)f0.x,(__bf16)f0.y,(__bf16)f0.z,(__bf16)f0.w,
                            (__bf16)f1.x,(__bf16)f1.y,(__bf16)f1.z,(__bf16)f1.w,
                            (__bf16)f2.x,(__bf16)f2.y,(__bf16)f2.z,(__bf16)f2.w,
                            (__bf16)f3.x,(__bf16)f3.y,(__bf16)f3.z,(__bf16)f3.w };
                v8bf blo = *(const v8bf*)(&afs[cn * AFS_STRIDE + kc + koffA]);
                v8bf bhi = *(const v8bf*)(&afs[cn * AFS_STRIDE + kc + 16 + koffA]);
                v16bf bb = __builtin_shufflevector(blo, bhi, 0,1,2,3,4,5,6,7,8,9,10,11,12,13,14,15);
                acc = WMMA_BF16(a, bb, acc);
            }
            #pragma unroll
            for (int r = 0; r < 8; ++r) {
                int row = mt * 16 + r + ((lane & 16) ? 8 : 0);
                Pg[row * CC + nt * 16 + (lane & 15)] = acc[r];
            }
        }
    }
    __syncthreads();

    // action_logit[b][c] = dot(af[b][c][:], am[c][:]) + ab[c]   (fp32)
    if (tid < CC) {
        const int c = tid;
        const float* x = af + ((size_t)b * CC + c) * HH;
        const float* w = am + (size_t)c * HH;
        float s = 0.f;
        for (int h = 0; h < HH; h += 4) {
            float4 xv = *(const float4*)(x + h);
            float4 wv = *(const float4*)(w + h);
            s += xv.x * wv.x + xv.y * wv.y + xv.z * wv.z + xv.w * wv.w;
        }
        action_out[b * CC + c] = s + ab[c];
    }

    // recipe_logit[b][r] = sum_c softmax_c(P0[r]) * P1[r][c] + rbias[r]
    // (recipe_attention_bias is constant over c -> drops out of softmax)
    if (tid < RR) {
        const int r = tid;
        const float* p0 = P + r * CC;
        const float* p1 = P + RPAD * CC + r * CC;
        float mx = -3.4e38f;
        for (int c = 0; c < CC; ++c) mx = fmaxf(mx, p0[c]);
        float den = 0.f, num = 0.f;
        for (int c = 0; c < CC; ++c) {
            float e = __expf(p0[c] - mx);
            den += e;
            num += e * p1[c];
        }
        recipe_out[b * RR + r] = num / den + rbias[r];
    }
}

// =====================================================================
extern "C" void kernel_launch(void* const* d_in, const int* in_sizes, int n_in,
                              void* d_out, int out_size, void* d_ws, size_t ws_size,
                              hipStream_t stream)
{
    const float* seq   = (const float*)d_in[0];
    const int*   slen  = (const int*)  d_in[1];
    const float* Wat   = (const float*)d_in[2];
    const float* bat   = (const float*)d_in[3];
    const float* am    = (const float*)d_in[4];
    const float* ab    = (const float*)d_in[5];
    const float* ram   = (const float*)d_in[6];
    const float* rab   = (const float*)d_in[7];   // constant over c: no softmax effect
    const float* rm    = (const float*)d_in[8];
    const float* rbias = (const float*)d_in[9];
    (void)rab; (void)in_sizes; (void)n_in; (void)out_size; (void)ws_size;

    // workspace layout (floats)
    float* ws     = (float*)d_ws;
    float* expl   = ws;                                  // [B,T,C] 4194304
    float* colsum = ws + (size_t)BB * TT * CC;           // [B,C]   2048
    float* af     = colsum + (size_t)BB * CC;            // [B,C,H] 1048576

    // output layout (floats): action_logit | recipe_logit | temporal_attention
    float* out_action = (float*)d_out;
    float* out_recipe = out_action + BB * CC;
    float* out_ta     = out_recipe + BB * RR;

    hipMemsetAsync(colsum, 0, (size_t)BB * CC * sizeof(float), stream);

    k1_gemm_exp<<<dim3(TT / 128, BB), 256, 0, stream>>>(seq, slen, Wat, bat, expl, colsum);

    k2_normalize<<<(BB * TT * CC) / (256 * 4), 256, 0, stream>>>(expl, colsum, out_ta);

    k3_action_feature<<<BB, 256, 0, stream>>>(seq, expl, colsum, af);

    const size_t smem4 = AFS_BYTES + P_FLOATS * sizeof(float);  // ~174 KB (<320 KB WGP LDS)
    k4_recipe<<<BB, 256, smem4, stream>>>(af, am, ab, ram, rm, rbias, out_action, out_recipe);
}
